// WithAttention_39273180955657
// MI455X (gfx1250) — matrix operand order repaired and checked by reading, not compiled, not run on hardware
//
#include <hip/hip_runtime.h>
#include <hip/hip_bf16.h>
#include <math.h>

// Problem sizes (match reference)
constexpr int cB = 64, cS = 1024, cH = 512, cA = 128, cD = 256;
constexpr int c3H = 3 * cH;          // 1536
constexpr int cKX = cD + cH;         // 768 (concat input width)
constexpr int LP  = 40;              // LDS row pitch in halves: 80 B, 16-B aligned

typedef __attribute__((ext_vector_type(16))) __bf16 v16bf;
typedef __attribute__((ext_vector_type(2)))  __bf16 v2bf;
typedef __attribute__((ext_vector_type(8)))  float  v8f;

// ---------------------------------------------------------------------------
// helpers
// ---------------------------------------------------------------------------
// Native f32->bf16 conversion (lets the backend use packed cvt instructions
// instead of a manual round-to-nearest-even bit sequence).
static __device__ __forceinline__ unsigned pack2(float a, float b) {
    v2bf v;
    v[0] = (__bf16)a;
    v[1] = (__bf16)b;
    return __builtin_bit_cast(unsigned, v);
}
static __device__ __forceinline__ float sigmoid_fast(float x) {
    return 1.0f / (1.0f + __expf(-x));
}
static __device__ __forceinline__ float tanh_fast(float x) {
    x = fminf(15.0f, fmaxf(-15.0f, x));
    float e = __expf(2.0f * x);
    return __fdividef(e - 1.0f, e + 1.0f);
}

union FragU { v16bf v; uint4 q[2]; };

// A-matrix (16x32 bf16) fragment gather from LDS tile (row pitch LP halves).
// Lane: row M = lane&15, hi = lane>>4.
//   VGPR i<4 : K = 2i + 8*hi ; VGPR i>=4 : K = 16 + 2(i-4) + 8*hi
// -> halves [8hi .. 8hi+7] and [16+8hi .. 23+8hi]: two 16-B aligned b128 loads.
static __device__ __forceinline__ v16bf gather_a(const unsigned short* tile,
                                                 int row, int hi) {
    FragU f;
    const unsigned short* r = tile + row * LP + 8 * hi;
    f.q[0] = *(const uint4*)(r);
    f.q[1] = *(const uint4*)(r + 16);
    return f.v;
}

// B-matrix (32x16 bf16) fragment gather. Lane: col N = lane&15, hi = lane>>4.
//   halves [8hi .. 8hi+7] and [16+8hi .. 23+8hi] (same K order as A): two b128 loads.
static __device__ __forceinline__ v16bf gather_b(const unsigned short* tile,
                                                 int n, int hi) {
    FragU f;
    const unsigned short* r = tile + n * LP + 8 * hi;
    f.q[0] = *(const uint4*)(r);
    f.q[1] = *(const uint4*)(r + 16);
    return f.v;
}

// stage 16 f32 -> 16 bf16 (two b128 stores) / 8 f32 -> 8 bf16 (one b128 store)
static __device__ __forceinline__ void store_bf16_16(unsigned short* dst,
        float4 a0, float4 a1, float4 a2, float4 a3) {
    uint4 q0, q1;
    q0.x = pack2(a0.x, a0.y); q0.y = pack2(a0.z, a0.w);
    q0.z = pack2(a1.x, a1.y); q0.w = pack2(a1.z, a1.w);
    q1.x = pack2(a2.x, a2.y); q1.y = pack2(a2.z, a2.w);
    q1.z = pack2(a3.x, a3.y); q1.w = pack2(a3.z, a3.w);
    ((uint4*)dst)[0] = q0;
    ((uint4*)dst)[1] = q1;
}
static __device__ __forceinline__ void store_bf16_8(unsigned short* dst,
        float4 x0, float4 x1) {
    uint4 q;
    q.x = pack2(x0.x, x0.y); q.y = pack2(x0.z, x0.w);
    q.z = pack2(x1.x, x1.y); q.w = pack2(x1.z, x1.w);
    *(uint4*)dst = q;
}

// ---------------------------------------------------------------------------
// Kernel 0: dec_proj[b,a] = states[b,:] . Wd[a,:] + bd[a]      [B,A]
// ---------------------------------------------------------------------------
__global__ __launch_bounds__(256)
void decproj_kernel(const float* __restrict__ states, const float* __restrict__ Wd,
                    const float* __restrict__ bd, float* __restrict__ decp) {
    int idx = blockIdx.x * 256 + threadIdx.x;
    if (idx >= cB * cA) return;
    int b = idx / cA, a = idx % cA;
    const float4* s = (const float4*)(states + (size_t)b * cH);
    const float4* w = (const float4*)(Wd + (size_t)a * cH);
    float acc = 0.f;
#pragma unroll 4
    for (int i = 0; i < cH / 4; ++i) {
        float4 sv = s[i], wv = w[i];
        acc += sv.x * wv.x + sv.y * wv.y + sv.z * wv.z + sv.w * wv.w;
    }
    decp[idx] = acc + bd[a];
}

// ---------------------------------------------------------------------------
// Kernel 1: fused scores.
//   C[m,n] = encoded[m,:] . We[n,:]      (M=B*S, N=A=128, K=H=512, bf16 WMMA)
//   score[m] = sum_n Wv[n]*tanh(C[m,n] + be[n] + dec_proj[b,n]) + bv
// Block = 256 thr (8 waves), tile 128x128, K step 32, double-buffered LDS,
// one barrier per K iteration. Wave tiling: 2 M-subs x 4 N-subs.
// ---------------------------------------------------------------------------
__global__ __launch_bounds__(256)
void scores_kernel(const float* __restrict__ enc, const float* __restrict__ We,
                   const float* __restrict__ be, const float* __restrict__ decp,
                   const float* __restrict__ Wv, const float* __restrict__ bv,
                   float* __restrict__ scores) {
    __shared__ __align__(16) unsigned short At[2][128 * LP];
    __shared__ __align__(16) unsigned short Bt[2][128 * LP];
    __shared__ float red[128 * 33];

    const int tid   = threadIdx.x;
    const int wave  = tid >> 5;
    const int lane  = tid & 31;
    const int l16   = lane & 15;
    const int hi    = lane >> 4;
    const int waveM = wave >> 1;          // 0..3 -> M-subs {2wm, 2wm+1}
    const int waveN = wave & 1;           // 0..1 -> N-subs {4wn .. 4wn+3}

    const int m_base = blockIdx.x * 128;  // 128 | S: tile stays in one batch
    const int b      = m_base / cS;

    const int lrow = tid >> 1;            // 0..127 (loader row)
    const int lcol = (tid & 1) * 16;      // 0 or 16

    const float* encRow = enc + (size_t)(m_base + lrow) * cH + lcol;
    const float* weRow  = We  + (size_t)lrow * cH + lcol;
    unsigned short* aDst = &At[0][0] + lrow * LP + lcol;
    unsigned short* bDst = &Bt[0][0] + lrow * LP + lcol;
    constexpr int BUFSTRIDE = 128 * LP;

    v8f acc[2][4] = {};

    // ---- prologue: stage slab 0 into buffer 0 ----
    {
        const float4* ga = (const float4*)encRow;
        const float4* gb = (const float4*)weRow;
        store_bf16_16(aDst, ga[0], ga[1], ga[2], ga[3]);
        store_bf16_16(bDst, gb[0], gb[1], gb[2], gb[3]);
    }
    __syncthreads();

    constexpr int NK = cH / 32;           // 16
#pragma unroll 2
    for (int it = 0; it < NK; ++it) {
        const int cur = it & 1;
        const bool nxt = (it + 1 < NK);
        float4 a0, a1, a2, a3, b0, b1, b2, b3;
        if (nxt) {                        // issue next-slab global loads early
            const float4* ga = (const float4*)(encRow + (it + 1) * 32);
            const float4* gb = (const float4*)(weRow  + (it + 1) * 32);
            a0 = ga[0]; a1 = ga[1]; a2 = ga[2]; a3 = ga[3];
            b0 = gb[0]; b1 = gb[1]; b2 = gb[2]; b3 = gb[3];
            if (it + 2 < NK)              // gfx1250 global_prefetch_b8
                __builtin_prefetch((const char*)(encRow + (it + 2) * 32), 0, 1);
        }
        // ---- gather all fragments up front, then 8 back-to-back WMMAs ----
        const unsigned short* Ac = &At[cur][0];
        const unsigned short* Bc = &Bt[cur][0];
        v16bf afr[2], bfr[4];
#pragma unroll
        for (int ms = 0; ms < 2; ++ms)
            afr[ms] = gather_a(Ac, (2 * waveM + ms) * 16 + l16, hi);
#pragma unroll
        for (int t = 0; t < 4; ++t)
            bfr[t] = gather_b(Bc, (4 * waveN + t) * 16 + l16, hi);
#pragma unroll
        for (int t = 0; t < 4; ++t)
#pragma unroll
            for (int ms = 0; ms < 2; ++ms)
                acc[ms][t] = __builtin_amdgcn_wmma_f32_16x16x32_bf16(
                    false, afr[ms], false, bfr[t], (short)0, acc[ms][t],
                    false, false);
        if (nxt) {                        // stage next slab into other buffer
            store_bf16_16(aDst + (cur ^ 1) * BUFSTRIDE, a0, a1, a2, a3);
            store_bf16_16(bDst + (cur ^ 1) * BUFSTRIDE, b0, b1, b2, b3);
        }
        __syncthreads();                  // single barrier per iteration
    }

    // ---- fused epilogue: tanh(C + be + dec) * Wv, reduce over n ----
    float addt[4], scl[4];
#pragma unroll
    for (int t = 0; t < 4; ++t) {
        int n = (4 * waveN + t) * 16 + l16;
        addt[t] = be[n] + decp[b * cA + n];
        scl[t]  = Wv[n];
    }
#pragma unroll
    for (int ms = 0; ms < 2; ++ms) {
        float part[8] = {0.f, 0.f, 0.f, 0.f, 0.f, 0.f, 0.f, 0.f};
#pragma unroll
        for (int t = 0; t < 4; ++t)
#pragma unroll
            for (int j = 0; j < 8; ++j)
                part[j] += tanh_fast(acc[ms][t][j] + addt[t]) * scl[t];
#pragma unroll
        for (int j = 0; j < 8; ++j)
            red[((2 * waveM + ms) * 16 + j + 8 * hi) * 33 + waveN * 16 + l16] = part[j];
    }
    __syncthreads();
    if (tid < 128) {
        float s = 0.f;
#pragma unroll
        for (int i = 0; i < 32; ++i) s += red[tid * 33 + i];
        scores[m_base + tid] = s + bv[0];
    }
}

// ---------------------------------------------------------------------------
// Kernel 2: softmax over S per batch row
// ---------------------------------------------------------------------------
__global__ __launch_bounds__(256)
void softmax_kernel(const float* __restrict__ scores, float* __restrict__ probs) {
    __shared__ float sd[256];
    const int b = blockIdx.x, tid = threadIdx.x;
    const float* row = scores + (size_t)b * cS;

    float m = -3.4e38f;
    for (int i = tid; i < cS; i += 256) m = fmaxf(m, row[i]);
    sd[tid] = m; __syncthreads();
    for (int off = 128; off > 0; off >>= 1) {
        if (tid < off) sd[tid] = fmaxf(sd[tid], sd[tid + off]);
        __syncthreads();
    }
    const float mx = sd[0];
    __syncthreads();

    float s = 0.f;
    for (int i = tid; i < cS; i += 256) s += __expf(row[i] - mx);
    sd[tid] = s; __syncthreads();
    for (int off = 128; off > 0; off >>= 1) {
        if (tid < off) sd[tid] += sd[tid + off];
        __syncthreads();
    }
    const float inv = 1.0f / sd[0];
    for (int i = tid; i < cS; i += 256)
        probs[(size_t)b * cS + i] = __expf(row[i] - mx) * inv;
}

// ---------------------------------------------------------------------------
// Kernel 3: attention[b,h] = sum_s probs[b,s] * encoded[b,s,h]
// ---------------------------------------------------------------------------
__global__ __launch_bounds__(128)
void attention_kernel(const float* __restrict__ probs, const float* __restrict__ enc,
                      float* __restrict__ attn) {
    __shared__ float p[cS];
    const int b = blockIdx.x, tid = threadIdx.x;
    for (int i = tid; i < cS; i += 128) p[i] = probs[(size_t)b * cS + i];
    __syncthreads();
    const int h = blockIdx.y * 128 + tid;
    const float* ep = enc + (size_t)b * cS * cH + h;
    float acc = 0.f;
#pragma unroll 4
    for (int s = 0; s < cS; ++s) acc = fmaf(p[s], ep[(size_t)s * cH], acc);
    attn[(size_t)b * cH + h] = acc;
}

// ---------------------------------------------------------------------------
// Kernels 4/5: GRU GEMMs, out[m,n] = X[m,:] . W[n,:] + bias[n]
//   M = 64 (single tile), N = 1536 (12 blocks of 128), K in {768, 512}.
//   X = virtual concat [X1 (K1 cols) | X2 (K-K1 cols)].
// Double-buffered bf16 WMMA; wave tiling 2 M-subs x 2 N-subs.
// ---------------------------------------------------------------------------
__global__ __launch_bounds__(256)
void gru_gemm_kernel(const float* __restrict__ X1, const float* __restrict__ X2,
                     int K1, int K, int N,
                     const float* __restrict__ W, const float* __restrict__ bias,
                     float* __restrict__ out) {
    __shared__ __align__(16) unsigned short At[2][64 * LP];
    __shared__ __align__(16) unsigned short Bt[2][128 * LP];

    const int tid   = threadIdx.x;
    const int wave  = tid >> 5;
    const int lane  = tid & 31;
    const int l16   = lane & 15;
    const int hi    = lane >> 4;
    const int waveM = wave >> 2;          // 0..1 -> M-subs {2wm, 2wm+1}
    const int waveN = wave & 3;           // 0..3 -> N-subs {2wn, 2wn+1}
    const int n_base = blockIdx.x * 128;

    const int arow = tid >> 2, acoff = (tid & 3) * 8;   // 64 rows x 32 cols
    const int brow = tid >> 1, bcoff = (tid & 1) * 16;  // 128 rows x 32 cols

    const int K2 = K - K1;
    unsigned short* aDst = &At[0][0] + arow * LP + acoff;
    unsigned short* bDst = &Bt[0][0] + brow * LP + bcoff;
    constexpr int ABUF = 64 * LP, BBUF = 128 * LP;

    v8f acc[2][2] = {};

    auto xptr = [&](int kg) -> const float* {
        return (kg < K1) ? (X1 + (size_t)arow * K1 + kg)
                         : (X2 + (size_t)arow * K2 + (kg - K1));
    };

    // ---- prologue ----
    {
        const float4* gx = (const float4*)xptr(acoff);
        const float4* gw = (const float4*)(W + (size_t)(n_base + brow) * K + bcoff);
        store_bf16_8(aDst, gx[0], gx[1]);
        store_bf16_16(bDst, gw[0], gw[1], gw[2], gw[3]);
    }
    __syncthreads();

    const int NK = K / 32;
#pragma unroll 2
    for (int it = 0; it < NK; ++it) {
        const int cur = it & 1;
        const bool nxt = (it + 1 < NK);
        float4 x0, x1, w0, w1, w2, w3;
        if (nxt) {
            const float4* gx = (const float4*)xptr((it + 1) * 32 + acoff);
            const float4* gw = (const float4*)(W + (size_t)(n_base + brow) * K
                                               + (it + 1) * 32 + bcoff);
            x0 = gx[0]; x1 = gx[1];
            w0 = gw[0]; w1 = gw[1]; w2 = gw[2]; w3 = gw[3];
        }
        const unsigned short* Ac = &At[cur][0];
        const unsigned short* Bc = &Bt[cur][0];
        v16bf afr[2], bfr[2];
#pragma unroll
        for (int ms = 0; ms < 2; ++ms)
            afr[ms] = gather_a(Ac, (2 * waveM + ms) * 16 + l16, hi);
#pragma unroll
        for (int ns = 0; ns < 2; ++ns)
            bfr[ns] = gather_b(Bc, (2 * waveN + ns) * 16 + l16, hi);
#pragma unroll
        for (int ms = 0; ms < 2; ++ms)
#pragma unroll
            for (int ns = 0; ns < 2; ++ns)
                acc[ms][ns] = __builtin_amdgcn_wmma_f32_16x16x32_bf16(
                    false, afr[ms], false, bfr[ns], (short)0, acc[ms][ns],
                    false, false);
        if (nxt) {
            store_bf16_8(aDst + (cur ^ 1) * ABUF, x0, x1);
            store_bf16_16(bDst + (cur ^ 1) * BBUF, w0, w1, w2, w3);
        }
        __syncthreads();
    }

#pragma unroll
    for (int ns = 0; ns < 2; ++ns) {
        const int n = n_base + (2 * waveN + ns) * 16 + l16;
        const float bb = bias[n];
#pragma unroll
        for (int ms = 0; ms < 2; ++ms)
#pragma unroll
            for (int j = 0; j < 8; ++j)
                out[(size_t)((2 * waveM + ms) * 16 + j + 8 * hi) * N + n] =
                    acc[ms][ns][j] + bb;
    }
}

// ---------------------------------------------------------------------------
// Kernel 6: GRU gate math
// ---------------------------------------------------------------------------
__global__ __launch_bounds__(256)
void gru_gate_kernel(const float* __restrict__ gi, const float* __restrict__ gh,
                     const float* __restrict__ states, float* __restrict__ out) {
    int idx = blockIdx.x * 256 + threadIdx.x;
    if (idx >= cB * cH) return;
    int b = idx >> 9, h = idx & (cH - 1);
    const float* gib = gi + (size_t)b * c3H;
    const float* ghb = gh + (size_t)b * c3H;
    float r = sigmoid_fast(gib[h]      + ghb[h]);
    float z = sigmoid_fast(gib[h + cH] + ghb[h + cH]);
    float n = tanh_fast(gib[h + 2 * cH] + r * ghb[h + 2 * cH]);
    out[idx] = (1.0f - z) * n + z * states[idx];
}

// ---------------------------------------------------------------------------
// launch
// ---------------------------------------------------------------------------
extern "C" void kernel_launch(void* const* d_in, const int* in_sizes, int n_in,
                              void* d_out, int out_size, void* d_ws, size_t ws_size,
                              hipStream_t stream) {
    const float* encoded = (const float*)d_in[0];
    const float* inputs  = (const float*)d_in[1];
    const float* states  = (const float*)d_in[2];
    const float* We      = (const float*)d_in[3];
    const float* be      = (const float*)d_in[4];
    const float* Wd      = (const float*)d_in[5];
    const float* bd      = (const float*)d_in[6];
    const float* Wv      = (const float*)d_in[7];
    const float* bv      = (const float*)d_in[8];
    const float* W_ih    = (const float*)d_in[9];
    const float* W_hh    = (const float*)d_in[10];
    const float* b_ih    = (const float*)d_in[11];
    const float* b_hh    = (const float*)d_in[12];
    float* out = (float*)d_out;

    // workspace layout (floats)
    float* ws     = (float*)d_ws;
    float* decp   = ws;                          // B*A      =  8192
    float* scores = decp   + cB * cA;            // B*S      = 65536
    float* probs  = scores + cB * cS;            // B*S      = 65536
    float* attn   = probs  + cB * cS;            // B*H      = 32768
    float* gi     = attn   + cB * cH;            // B*3H     = 98304
    float* gh     = gi     + cB * c3H;           // B*3H     = 98304

    decproj_kernel<<<(cB * cA + 255) / 256, 256, 0, stream>>>(states, Wd, bd, decp);
    scores_kernel<<<cB * cS / 128, 256, 0, stream>>>(encoded, We, be, decp, Wv, bv, scores);
    softmax_kernel<<<cB, 256, 0, stream>>>(scores, probs);
    attention_kernel<<<dim3(cB, cH / 128), 128, 0, stream>>>(probs, encoded, attn);
    gru_gemm_kernel<<<c3H / 128, 256, 0, stream>>>(inputs, attn, cD, cKX, c3H, W_ih, b_ih, gi);
    gru_gemm_kernel<<<c3H / 128, 256, 0, stream>>>(states, states, cH, cH, c3H, W_hh, b_hh, gh);
    gru_gate_kernel<<<(cB * cH + 255) / 256, 256, 0, stream>>>(gi, gh, states, out);
}
